// VideoSAGE_88648124989745
// MI455X (gfx1250) — compile-verified, matrix operand order checked by reading, not run on hardware
//
#include <hip/hip_runtime.h>
#include <hip/hip_bf16.h>

typedef __bf16 bf16_t;
typedef __attribute__((ext_vector_type(16))) __bf16 v16bf;
typedef __attribute__((ext_vector_type(8)))  __bf16 v8bf;
typedef __attribute__((ext_vector_type(8)))  float  v8f;

#define BB    8
#define SS    2048
#define FIN   1024
#define HH    512
#define SC1   256
#define SC2   128
#define MROWS (BB*SS)   // 16384

// Load one 16x32 bf16 WMMA fragment row (A) or column-run (B from transposed
// weights). Per CDNA5 ISA 16-bit A layout: lanes 0-15 hold K = {0..7,16..23},
// lanes 16-31 hold K = {8..15,24..31} for row M = lane&15. Two aligned 16-byte
// contiguous loads per lane.
__device__ __forceinline__ v16bf load_frag16(const bf16_t* __restrict__ p, int half) {
  v8bf lo = *(const v8bf*)(p + half * 8);
  v8bf hi = *(const v8bf*)(p + 16 + half * 8);
  return __builtin_shufflevector(lo, hi, 0,1,2,3,4,5,6,7,8,9,10,11,12,13,14,15);
}

// C[M,N] = A[M,K] (bf16 row-major) * Bt[N,K]^T (bf16, weights pre-transposed)
// + bias, optional ReLU. Output bf16 (feeds next GEMM) or fp32 (feeds stencil).
// One wave computes a 16x64 strip: A fragment reused across 4 WMMAs per K-step.
__global__ __launch_bounds__(256)
void gemm_bf16_wmma(const bf16_t* __restrict__ A,
                    const bf16_t* __restrict__ Bt,
                    const float*  __restrict__ bias,
                    bf16_t* __restrict__ Cb,
                    float*  __restrict__ Cf,
                    int M, int N, int K, int relu, int out_f32)
{
  const int lane = threadIdx.x & 31;
  const int wave = threadIdx.x >> 5;
  const int half = lane >> 4;
  const int l16  = lane & 15;
  const int m0 = blockIdx.y * 128 + wave * 16;
  const int n0 = blockIdx.x * 64;
  if (m0 >= M) return;  // wave-uniform; EXEC stays all-ones for WMMA

  const bf16_t* __restrict__ arow = A  + (size_t)(m0 + l16) * K;
  const bf16_t* __restrict__ brow = Bt + (size_t)(n0 + l16) * K;

  v8f c[4] = {};
  for (int kb = 0; kb < K; kb += 32) {
    v16bf a = load_frag16(arow + kb, half);
#pragma unroll
    for (int sub = 0; sub < 4; ++sub) {
      v16bf b = load_frag16(brow + (size_t)sub * 16 * K + kb, half);
      c[sub] = __builtin_amdgcn_wmma_f32_16x16x32_bf16(
          /*neg_a=*/false, a, /*neg_b=*/false, b,
          /*c_mod=*/(short)0, c[sub], /*reuse_a=*/false, /*reuse_b=*/false);
    }
  }

  // C/D layout: lane = N column (l16), VGPR r -> M = r + 8*half
#pragma unroll
  for (int sub = 0; sub < 4; ++sub) {
    const int n = n0 + sub * 16 + l16;
    const float bv = bias ? bias[n] : 0.0f;
#pragma unroll
    for (int r = 0; r < 8; ++r) {
      const int m = m0 + half * 8 + r;
      float v = c[sub][r] + bv;
      if (relu) v = fmaxf(v, 0.0f);
      if (out_f32) Cf[(size_t)m * N + n] = v;
      else         Cb[(size_t)m * N + n] = (bf16_t)v;
    }
  }
}

// h[b,s,k] = dinv[s] * sum_{t in {s-1,s,s+1}} dinv[t]*sup[b,t,k] + bias[k]
// (tridiagonal normalized adjacency applied as a stencil; deg=2 at sequence
// ends, 3 inside). Writes bf16 for the next GEMM.
__global__ void gcn_stencil(const float* __restrict__ sup,
                            const float* __restrict__ bias,
                            bf16_t* __restrict__ hout,
                            int relu, size_t total)
{
  size_t idx = (size_t)blockIdx.x * blockDim.x + threadIdx.x;
  if (idx >= total) return;
  const int    k   = (int)(idx % HH);
  const size_t row = idx / HH;
  const int    s   = (int)(row % SS);

  const float dself = rsqrtf(((s == 0) || (s == SS - 1) ? 2.0f : 3.0f) + 1e-8f);
  float acc = dself * sup[idx];
  if (s > 0) {
    const float dm = rsqrtf(((s - 1 == 0) ? 2.0f : 3.0f) + 1e-8f);
    acc += dm * sup[idx - HH];
  }
  if (s < SS - 1) {
    const float dp = rsqrtf(((s + 1 == SS - 1) ? 2.0f : 3.0f) + 1e-8f);
    acc += dp * sup[idx + HH];
  }
  float v = dself * acc + bias[k];
  if (relu) v = fmaxf(v, 0.0f);
  hout[idx] = (bf16_t)v;
}

// One wave (32 lanes) per output row: 128-wide dot product + sigmoid.
__global__ void score_sigmoid(const bf16_t* __restrict__ z2,
                              const float* __restrict__ w,
                              const float* __restrict__ b,
                              float* __restrict__ out)
{
  const int row  = blockIdx.x;
  const int lane = threadIdx.x;
  const bf16_t* __restrict__ zr = z2 + (size_t)row * SC2;
  float acc = 0.0f;
#pragma unroll
  for (int d = lane; d < SC2; d += 32) acc += (float)zr[d] * w[d];
#pragma unroll
  for (int off = 16; off > 0; off >>= 1) acc += __shfl_xor(acc, off, 32);
  if (lane == 0) out[row] = 1.0f / (1.0f + expf(-(acc + b[0])));
}

__global__ void cast_f32_bf16(const float* __restrict__ in, bf16_t* __restrict__ out, size_t n) {
  size_t i = (size_t)blockIdx.x * blockDim.x + threadIdx.x;
  const size_t stride = (size_t)gridDim.x * blockDim.x;
  for (; i < n; i += stride) out[i] = (bf16_t)in[i];
}

// fp32 [K][N] -> bf16 [N][K] so WMMA B-fragments load as contiguous K-runs.
__global__ void transpose_f32_bf16(const float* __restrict__ in, bf16_t* __restrict__ out,
                                   int K, int N) {
  size_t i = (size_t)blockIdx.x * blockDim.x + threadIdx.x;
  if (i >= (size_t)K * N) return;
  const int n = (int)(i % N);
  const int k = (int)(i / N);
  out[(size_t)n * K + k] = (bf16_t)in[i];
}

extern "C" void kernel_launch(void* const* d_in, const int* in_sizes, int n_in,
                              void* d_out, int out_size, void* d_ws, size_t ws_size,
                              hipStream_t stream) {
  const float* x   = (const float*)d_in[0];
  const float* Wp  = (const float*)d_in[1];
  const float* bp  = (const float*)d_in[2];
  const float* gW  = (const float*)d_in[3];
  const float* gb  = (const float*)d_in[4];
  const float* s1W = (const float*)d_in[5];
  const float* s1b = (const float*)d_in[6];
  const float* s2W = (const float*)d_in[7];
  const float* s2b = (const float*)d_in[8];
  const float* s3W = (const float*)d_in[9];
  const float* s3b = (const float*)d_in[10];
  float* out = (float*)d_out;

  char* ws = (char*)d_ws;
  auto carve = [&](size_t bytes) -> char* {
    char* p = ws;
    ws += (bytes + 255) & ~(size_t)255;
    return p;
  };
  bf16_t* xb  = (bf16_t*)carve((size_t)MROWS * FIN * 2);  // 32 MB
  bf16_t* hb  = (bf16_t*)carve((size_t)MROWS * HH  * 2);  // 16 MB
  float*  sup = (float*) carve((size_t)MROWS * HH  * 4);  // 32 MB
  bf16_t* z1  = (bf16_t*)carve((size_t)MROWS * SC1 * 2);  //  8 MB
  bf16_t* z2  = (bf16_t*)carve((size_t)MROWS * SC2 * 2);  //  4 MB
  bf16_t* WpT = (bf16_t*)carve((size_t)FIN * HH * 2);
  bf16_t* gWT = (bf16_t*)carve((size_t)3 * HH * HH * 2);
  bf16_t* s1T = (bf16_t*)carve((size_t)HH * SC1 * 2);
  bf16_t* s2T = (bf16_t*)carve((size_t)SC1 * SC2 * 2);

  // --- one-off conversions ---
  cast_f32_bf16<<<4096, 256, 0, stream>>>(x, xb, (size_t)MROWS * FIN);
  transpose_f32_bf16<<<(FIN * HH + 255) / 256, 256, 0, stream>>>(Wp, WpT, FIN, HH);
  for (int i = 0; i < 3; ++i)
    transpose_f32_bf16<<<(HH * HH + 255) / 256, 256, 0, stream>>>(
        gW + (size_t)i * HH * HH, gWT + (size_t)i * HH * HH, HH, HH);
  transpose_f32_bf16<<<(HH * SC1 + 255) / 256, 256, 0, stream>>>(s1W, s1T, HH, SC1);
  transpose_f32_bf16<<<(SC1 * SC2 + 255) / 256, 256, 0, stream>>>(s2W, s2T, SC1, SC2);

  // --- projection: h = x @ Wp + bp (bf16 out) ---
  gemm_bf16_wmma<<<dim3(HH / 64, MROWS / 128), 256, 0, stream>>>(
      xb, WpT, bp, hb, nullptr, MROWS, HH, FIN, /*relu=*/0, /*out_f32=*/0);

  // --- 3 GCN layers: GEMM (fp32 out) then tridiagonal-adjacency stencil ---
  const size_t total = (size_t)MROWS * HH;
  for (int i = 0; i < 3; ++i) {
    gemm_bf16_wmma<<<dim3(HH / 64, MROWS / 128), 256, 0, stream>>>(
        hb, gWT + (size_t)i * HH * HH, nullptr, nullptr, sup,
        MROWS, HH, HH, /*relu=*/0, /*out_f32=*/1);
    gcn_stencil<<<(unsigned)((total + 255) / 256), 256, 0, stream>>>(
        sup, gb + (size_t)i * HH, hb, /*relu=*/(i < 2) ? 1 : 0, total);
  }

  // --- scorer MLP ---
  gemm_bf16_wmma<<<dim3(SC1 / 64, MROWS / 128), 256, 0, stream>>>(
      hb, s1T, s1b, z1, nullptr, MROWS, SC1, HH, /*relu=*/1, /*out_f32=*/0);
  gemm_bf16_wmma<<<dim3(SC2 / 64, MROWS / 128), 256, 0, stream>>>(
      z1, s2T, s2b, z2, nullptr, MROWS, SC2, SC1, /*relu=*/1, /*out_f32=*/0);
  score_sigmoid<<<MROWS, 32, 0, stream>>>(z2, s3W, s3b, out);
}
